// LoRALayer_69329362092463
// MI455X (gfx1250) — matrix operand order repaired and checked by reading, not compiled, not run on hardware
//
#include <hip/hip_runtime.h>

// ---- CDNA5 WMMA vector types ----
typedef __attribute__((ext_vector_type(16))) __bf16 v16bf;
typedef __attribute__((ext_vector_type(8)))  __bf16 v8bf;
typedef __attribute__((ext_vector_type(2)))  __bf16 v2bf;
typedef __attribute__((ext_vector_type(8)))  float  v8f;
typedef __attribute__((ext_vector_type(4)))  unsigned short us4;

// Problem constants (all tile sizes divide evenly -> no bounds checks)
constexpr int Dd   = 4096;   // reduction dim
constexpr int Nn   = 4096;   // output features
constexpr int Mm   = 8192;   // 4 * 2048 rows
constexpr int RANK = 16;
constexpr float LSCALE = 2.0f;   // ALPHA/RANK = 32/16

constexpr int BM  = 128;
constexpr int BN  = 128;
constexpr int BK  = 64;
constexpr int LDK = 72;      // padded LDS row stride (bf16 elems); 144B rows, 16B-aligned

// fp32 -> bf16 bits, round-to-nearest-even (scalar fallback)
static __device__ __forceinline__ unsigned short f2bf(float f) {
    unsigned u = __float_as_uint(f);
    return (unsigned short)((u + 0x7FFFu + ((u >> 16) & 1u)) >> 16);
}
static __device__ __forceinline__ us4 pack4(float4 v) {
#if __has_builtin(__builtin_amdgcn_cvt_pk_bf16_f32)
    v2bf lo = __builtin_amdgcn_cvt_pk_bf16_f32(v.x, v.y);
    v2bf hi = __builtin_amdgcn_cvt_pk_bf16_f32(v.z, v.w);
    us4 r;
    __builtin_memcpy(&r, &lo, 4);
    __builtin_memcpy((char*)&r + 4, &hi, 4);
    return r;
#else
    us4 r; r.x = f2bf(v.x); r.y = f2bf(v.y); r.z = f2bf(v.z); r.w = f2bf(v.w);
    return r;
#endif
}
// Load a 16-bf16 fragment as two 16B-aligned LDS reads; second run at +hoff elems
static __device__ __forceinline__ v16bf frag2(const unsigned short* p, int hoff) {
    v8bf lo = *(const v8bf*)(p);
    v8bf hi = *(const v8bf*)(p + hoff);
    return __builtin_shufflevector(lo, hi, 0,1,2,3,4,5,6,7,8,9,10,11,12,13,14,15);
}

__global__ __launch_bounds__(256) void lora_fused_gemm(
    const float* __restrict__ x,    // [Mm, Dd]
    const float* __restrict__ W,    // [Nn, Dd]
    const float* __restrict__ bias, // [Nn]
    const float* __restrict__ Bm,   // [Dd, RANK]
    const float* __restrict__ Am,   // [RANK, Nn]
    float* __restrict__ out)        // [Mm, Nn]
{
    // Double-buffered bf16 tiles
    __shared__ unsigned short xs [2][BM * LDK];
    __shared__ unsigned short wsd[2][BN * LDK];
    __shared__ unsigned short bst[2][RANK * LDK];  // LoRA-B chunk, transposed [r][k]
    __shared__ float tl[BM * RANK];                // t = x@B tile (f32)
    __shared__ float al[RANK * BN];                // A tile (f32)
    __shared__ float bl[BN];                       // bias tile

    const int tid     = threadIdx.x;
    const int lane    = tid & 31;
    const int wave    = tid >> 5;
    const int lane_lo = lane & 15;
    const int lane_hi = lane >> 4;          // 0 or 1
    const int wave_m  = wave & 3;           // 4 M-rows of waves
    const int wave_n  = wave >> 2;          // 2 N-cols of waves
    const int gm0 = blockIdx.y * BM;
    const int gn0 = blockIdx.x * BN;

    // Cooperative-loader indices: 16 float4 per row, 16 base rows, 8 rows/thread
    const int lc4 = tid & 15;               // float4 column
    const int lr0 = tid >> 4;               // base row
    const int br  = tid & 15;               // LoRA-B: rank index
    const int bk0 = tid >> 4;               // LoRA-B: base k

    // Per-lane fragment bases (ISA 7.12.2 layouts)
    const int rowA0 = (wave_m * 32 + lane_lo) * LDK;              // A frag sub 0
    const int rowAL = (wave_m * 32 + wave_n * 16 + lane_lo) * LDK;// LoRA A frag (this wave's sub)
    const int rowB0 = (wave_n * 64 + lane_lo) * LDK;              // B frag sub 0
    const int khA   = lane_hi * 8;                                // A: K runs at khA, khA+16
    const int kB    = lane_hi * 8;                                // B: K runs at kB, kB+16
    const int rowR  = lane_lo * LDK;                              // bst: r = lane_lo

    v8f acc[2][4];
    #pragma unroll
    for (int i = 0; i < 2; ++i)
        #pragma unroll
        for (int j = 0; j < 4; ++j) acc[i][j] = v8f{};
    v8f tacc = v8f{};

    // Register staging for double buffering
    float4 sx[8], sw[8];
    float  sb[4];

    // ---- prologue: fill buffer 0 ----
    #pragma unroll
    for (int i = 0; i < 8; ++i) {
        const int row = lr0 + 16 * i;
        sx[i] = *(const float4*)(x + (size_t)(gm0 + row) * Dd + lc4 * 4);
        sw[i] = *(const float4*)(W + (size_t)(gn0 + row) * Dd + lc4 * 4);
    }
    #pragma unroll
    for (int i = 0; i < 4; ++i)
        sb[i] = Bm[(size_t)(bk0 + 16 * i) * RANK + br];
    #pragma unroll
    for (int i = 0; i < 8; ++i) {
        const int row = lr0 + 16 * i;
        *(us4*)(&xs [0][row * LDK + lc4 * 4]) = pack4(sx[i]);
        *(us4*)(&wsd[0][row * LDK + lc4 * 4]) = pack4(sw[i]);
    }
    #pragma unroll
    for (int i = 0; i < 4; ++i)
        bst[0][br * LDK + bk0 + 16 * i] = f2bf(sb[i]);
    __syncthreads();

    for (int kt = 0; kt < Dd; kt += BK) {
        const int  cur = (kt >> 6) & 1;
        const bool have_next = (kt + BK) < Dd;

        // ---- issue next chunk's global loads (independent of this chunk's math) ----
        if (have_next) {
            const int kn = kt + BK;
            #pragma unroll
            for (int i = 0; i < 8; ++i) {
                const int row = lr0 + 16 * i;
                sx[i] = *(const float4*)(x + (size_t)(gm0 + row) * Dd + kn + lc4 * 4);
                sw[i] = *(const float4*)(W + (size_t)(gn0 + row) * Dd + kn + lc4 * 4);
            }
            #pragma unroll
            for (int i = 0; i < 4; ++i)
                sb[i] = Bm[(size_t)(kn + bk0 + 16 * i) * RANK + br];
        }

        // ---- 2 WMMA K-steps on current buffer ----
        #pragma unroll
        for (int ks = 0; ks < BK; ks += 32) {
            v16bf afr[2], bfr[4];
            #pragma unroll
            for (int i = 0; i < 2; ++i)
                afr[i] = frag2(&xs[cur][rowA0 + i * 16 * LDK + ks + khA], 16);
            #pragma unroll
            for (int j = 0; j < 4; ++j)
                bfr[j] = frag2(&wsd[cur][rowB0 + j * 16 * LDK + ks + kB], 16);
            const v16bf alo = frag2(&xs[cur][rowAL + ks + khA], 16);  // LoRA A frag (no cndmask)
            const v16bf bbf = frag2(&bst[cur][rowR + ks + kB], 16);

            #pragma unroll
            for (int i = 0; i < 2; ++i)
                #pragma unroll
                for (int j = 0; j < 4; ++j)
                    acc[i][j] = __builtin_amdgcn_wmma_f32_16x16x32_bf16(
                        false, afr[i], false, bfr[j], (short)0, acc[i][j], false, false);

            tacc = __builtin_amdgcn_wmma_f32_16x16x32_bf16(
                false, alo, false, bbf, (short)0, tacc, false, false);
        }

        // ---- convert + store next chunk into the other buffer ----
        if (have_next) {
            const int nxt = cur ^ 1;
            #pragma unroll
            for (int i = 0; i < 8; ++i) {
                const int row = lr0 + 16 * i;
                *(us4*)(&xs [nxt][row * LDK + lc4 * 4]) = pack4(sx[i]);
                *(us4*)(&wsd[nxt][row * LDK + lc4 * 4]) = pack4(sw[i]);
            }
            #pragma unroll
            for (int i = 0; i < 4; ++i)
                bst[nxt][br * LDK + bk0 + 16 * i] = f2bf(sb[i]);
        }
        __syncthreads();
    }

    // ---- epilogue staging: t -> LDS (C-layout: VGPR j is M=j+8*lane_hi, N=lane_lo) ----
    {
        const int mtb = wave_m * 32 + wave_n * 16;
        #pragma unroll
        for (int j = 0; j < 8; ++j)
            tl[(mtb + j + 8 * lane_hi) * RANK + lane_lo] = tacc[j];
    }
    #pragma unroll
    for (int i = 0; i < 2; ++i) {      // A tile [16][128]
        const int idx = tid + 256 * i;
        const int r = idx >> 5, c4 = idx & 31;
        *(float4*)(al + r * BN + c4 * 4) = *(const float4*)(Am + (size_t)r * Nn + gn0 + c4 * 4);
    }
    if (tid < 32)
        *(float4*)(bl + tid * 4) = *(const float4*)(bias + gn0 + tid * 4);
    __syncthreads();

    // ---- epilogue: out = acc + bias + LSCALE * (t @ A) ----
    #pragma unroll
    for (int sn = 0; sn < 4; ++sn) {
        const int n_loc = wave_n * 64 + sn * 16 + lane_lo;
        const float bv = bl[n_loc];
        float ar[RANK];
        #pragma unroll
        for (int r = 0; r < RANK; ++r) ar[r] = al[r * BN + n_loc];
        #pragma unroll
        for (int sm = 0; sm < 2; ++sm) {
            #pragma unroll
            for (int j = 0; j < 8; ++j) {
                const int m_loc = wave_m * 32 + sm * 16 + j + 8 * lane_hi;
                float l = 0.f;
                #pragma unroll
                for (int r = 0; r < RANK; ++r)
                    l = fmaf(tl[m_loc * RANK + r], ar[r], l);
                out[(size_t)(gm0 + m_loc) * Nn + gn0 + n_loc] =
                    acc[sm][sn][j] + bv + LSCALE * l;
            }
        }
    }
}

extern "C" void kernel_launch(void* const* d_in, const int* in_sizes, int n_in,
                              void* d_out, int out_size, void* d_ws, size_t ws_size,
                              hipStream_t stream) {
    (void)in_sizes; (void)n_in; (void)d_ws; (void)ws_size; (void)out_size;
    const float* x    = (const float*)d_in[0];
    const float* W    = (const float*)d_in[1];
    const float* bias = (const float*)d_in[2];
    const float* Bm   = (const float*)d_in[3];
    const float* Am   = (const float*)d_in[4];
    float* out = (float*)d_out;

    dim3 grid(Nn / BN, Mm / BM);   // 32 x 64 = 2048 blocks
    dim3 block(256);               // 8 waves (wave32)
    lora_fused_gemm<<<grid, block, 0, stream>>>(x, W, bias, Bm, Am, out);
}
